// SpiralDeblock_2808908612152
// MI455X (gfx1250) — matrix operand order repaired and checked by reading, not compiled, not run on hardware
//
#include <hip/hip_runtime.h>
#include <hip/hip_bf16.h>

typedef _Float16 v16h __attribute__((ext_vector_type(16)));
typedef _Float16 v8h  __attribute__((ext_vector_type(8)));
typedef float    v8f  __attribute__((ext_vector_type(8)));

constexpr int B_    = 4;
constexpr int V_IN  = 12500;
constexpr int V_OUT = 50000;
constexpr int C_IN  = 64;
constexpr int C_OUT = 32;
constexpr int SPIRAL = 9;
constexpr int K_NNZ  = 3;
constexpr int KF     = SPIRAL * C_IN;      // 576
constexpr int NTILES = B_ * (V_OUT / 16);  // 12500 wave-tiles
constexpr int TILES_PER_BLOCK = 8;         // 8 waves * 32 lanes = 256 threads
constexpr size_t UP_BYTES = (size_t)B_ * V_OUT * C_IN * sizeof(_Float16); // 25.6 MB
constexpr int W_BYTES = KF * C_OUT * sizeof(_Float16);                    // 36864 B

// ---------------------------------------------------------------------------
// Kernel 0: one-time weight convert f32 [k][n] row-major -> f16 [n][k]
// column-major in workspace, so per-block staging is a pure linear memcpy.
// ---------------------------------------------------------------------------
__global__ __launch_bounds__(256) void convert_weight_kernel(
    const float* __restrict__ weight, _Float16* __restrict__ wh) {
  int i = blockIdx.x * 256 + threadIdx.x;     // exact: KF*C_OUT threads
  int k = i / C_OUT, n = i % C_OUT;
  wh[n * KF + k] = (_Float16)weight[i];
}

// ---------------------------------------------------------------------------
// Kernel 1: sparse upsample  up[b,v,c] = sum_k x[b, up_idx[v,k], c]*up_val[v,k]
// stored as f16 into workspace (halves gather traffic for the GEMM).
// ---------------------------------------------------------------------------
__global__ __launch_bounds__(256) void upsample_f16_kernel(
    const float* __restrict__ x, const int* __restrict__ up_idx,
    const float* __restrict__ up_val, _Float16* __restrict__ up_h) {
  int idx = blockIdx.x * 256 + threadIdx.x;     // exact: B*V_OUT*64 threads
  int c = idx & (C_IN - 1);
  int v = (idx >> 6) % V_OUT;
  int b = idx / (V_OUT * C_IN);
  float acc = 0.f;
#pragma unroll
  for (int k = 0; k < K_NNZ; ++k) {
    int r = up_idx[v * K_NNZ + k];
    acc += x[((size_t)b * V_IN + r) * C_IN + c] * up_val[v * K_NNZ + k];
  }
  up_h[(size_t)idx] = (_Float16)acc;
}

// ---------------------------------------------------------------------------
// Kernel 2: spiral gather + 16x576 @ 576x32 GEMM per wave via
// v_wmma_f32_16x16x32_f16.  One wave32 owns a 16-vertex x 32-output tile.
// Weight tile (36 KB, already f16 column-major) is staged into LDS with the
// CDNA5 async global->LDS path (ASYNCcnt) instead of a VGPR round-trip.
// ---------------------------------------------------------------------------
__global__ __launch_bounds__(256) void spiral_gemm_kernel(
    const _Float16* __restrict__ up_h, const int* __restrict__ spiral,
    const _Float16* __restrict__ wh, const float* __restrict__ bias,
    float* __restrict__ out) {
  __shared__ _Float16 wlds[C_OUT * KF];       // column-major: wlds[n*576 + k], 36 KB

  // Async-stage weight: 9 x b128 per thread; 256 threads * 16 B * 9 = 36864 B.
  {
    const char* wsrc = (const char*)wh;
    uint32_t lbase = (uint32_t)(uintptr_t)(&wlds[0]);   // low 32 bits = LDS offset
    uint32_t toff = (uint32_t)threadIdx.x * 16u;
#pragma unroll
    for (int it = 0; it < W_BYTES / (256 * 16); ++it) {
      uint32_t boff = (uint32_t)it * 4096u + toff;
      uint64_t ga = (uint64_t)(uintptr_t)(wsrc + boff);
      uint32_t la = lbase + boff;
      asm volatile("global_load_async_to_lds_b128 %0, %1, off"
                   :: "v"(la), "v"(ga) : "memory");
    }
    asm volatile("s_wait_asynccnt 0x0" ::: "memory");
  }
  __syncthreads();

  int wave = threadIdx.x >> 5;
  int lane = threadIdx.x & 31;
  int tile = blockIdx.x * TILES_PER_BLOCK + wave;
  if (tile >= NTILES) return;                 // wave-uniform: EXEC stays all-1s

  int b  = tile / (V_OUT / 16);
  int v0 = (tile % (V_OUT / 16)) * 16;
  int m  = lane & 15;
  int hi = lane >> 4;
  const _Float16* upb = up_h + (size_t)b * V_OUT * C_IN;

  // Bias pre-loaded into accumulators: each lane owns one output column.
  float b0 = bias[m];
  float b1 = bias[16 + m];
  v8f acc0, acc1;
#pragma unroll
  for (int i = 0; i < 8; ++i) { acc0[i] = b0; acc1[i] = b1; }

#pragma unroll 2
  for (int t = 0; t < KF / 32; ++t) {         // 18 K-chunks of 32
    int s     = t >> 1;
    int cbase = (t & 1) * 32;
    int rv    = spiral[(v0 + m) * SPIRAL + s];
    const _Float16* rowp = upb + (size_t)rv * C_IN;
    int c0 = cbase + hi * 16;
    v8h alo = *(const v8h*)(rowp + c0);       // K-chunk halves 0..7
    v8h ahi = *(const v8h*)(rowp + c0 + 8);   // K-chunk halves 8..15
    v16h a;
#pragma unroll
    for (int i = 0; i < 8; ++i) { a[i] = alo[i]; a[8 + i] = ahi[i]; }

    int k0 = t * 32 + hi * 16;
    v16h bm0 = *(const v16h*)&wlds[(size_t)m * KF + k0];
    v16h bm1 = *(const v16h*)&wlds[(size_t)(16 + m) * KF + k0];

    acc0 = __builtin_amdgcn_wmma_f32_16x16x32_f16(false, a, false, bm0,
                                                  (short)0, acc0, false, false);
    acc1 = __builtin_amdgcn_wmma_f32_16x16x32_f16(false, a, false, bm1,
                                                  (short)0, acc1, false, false);
  }

  // D layout: VGPR r -> (M = r + hi*8, N = lane&15) for each 16-wide N tile.
  float* outb = out + ((size_t)b * V_OUT + v0) * C_OUT;
#pragma unroll
  for (int r = 0; r < 8; ++r) {
    int row = r + hi * 8;
    float y0 = acc0[r]; y0 = y0 > 0.f ? y0 : 0.f;
    float y1 = acc1[r]; y1 = y1 > 0.f ? y1 : 0.f;
    outb[(size_t)row * C_OUT + m]      = y0;
    outb[(size_t)row * C_OUT + 16 + m] = y1;
  }
}

extern "C" void kernel_launch(void* const* d_in, const int* in_sizes, int n_in,
                              void* d_out, int out_size, void* d_ws, size_t ws_size,
                              hipStream_t stream) {
  const float* x        = (const float*)d_in[0];
  const int*   spiral   = (const int*)d_in[1];
  const int*   up_idx   = (const int*)d_in[2];
  const float* up_val   = (const float*)d_in[3];
  const float* weight   = (const float*)d_in[4];
  const float* bias     = (const float*)d_in[5];
  float*       y        = (float*)d_out;
  _Float16*    up_h     = (_Float16*)d_ws;                       // 25.6 MB
  _Float16*    wh       = (_Float16*)((char*)d_ws + UP_BYTES);   // 36 KB, 16B-aligned

  // Kernel 0: one-time weight convert (18432 elements).
  convert_weight_kernel<<<(KF * C_OUT) / 256, 256, 0, stream>>>(weight, wh);

  // Kernel 1: exact grid, B*V_OUT*C_IN = 12.8M threads.
  upsample_f16_kernel<<<(B_ * V_OUT * C_IN) / 256, 256, 0, stream>>>(
      x, up_idx, up_val, up_h);

  // Kernel 2: 12500 wave-tiles, 8 per block.
  int blocks = (NTILES + TILES_PER_BLOCK - 1) / TILES_PER_BLOCK;
  spiral_gemm_kernel<<<blocks, 256, 0, stream>>>(up_h, spiral, wh, bias, y);
}
